// Gpt2Attention_25735444038402
// MI455X (gfx1250) — matrix operand order repaired and checked by reading, not compiled, not run on hardware
//
#include <hip/hip_runtime.h>
#include <hip/hip_bf16.h>

// GPT-2 attention, B=2, S=2048, E=1024, H=16, D=64, all bf16-WMMA on gfx1250.

#define SEQ    2048
#define NEMBD  1024
#define NHEAD  16
#define HDIM   64
#define LDQKV  3072
#define MROWS  4096          // B*S
#define NEGV  -1.0e9f

typedef __attribute__((ext_vector_type(16))) __bf16 v16bf;
typedef __attribute__((ext_vector_type(8)))  float  v8f;

__device__ __forceinline__ v8f wmma_bf16(v16bf a, v16bf b, v8f c) {
  // (neg_a, A, neg_b, B, c_mod, C, reuse_a, reuse_b)
  return __builtin_amdgcn_wmma_f32_16x16x32_bf16(false, a, false, b, (short)0, c,
                                                 false, false);
}

// Async global->LDS copy, 128b per lane (ASYNCcnt-tracked).  ldsOff is the
// block-relative LDS byte address.
__device__ __forceinline__ void async_ld128(unsigned ldsOff, const float* g) {
  asm volatile("global_load_async_to_lds_b128 %0, %1, off"
               :: "v"(ldsOff), "v"(g) : "memory");
}
__device__ __forceinline__ void wait_async_0() {
  asm volatile("s_wait_asynccnt 0x0" ::: "memory");
}
__device__ __forceinline__ void wait_async_16() {
  asm volatile("s_wait_asynccnt 0x10" ::: "memory");
}

// ---------------------------------------------------------------------------
// C[M,N] = X[M,K] * W[N,K]^T + bias[N]    (Conv1D / x @ w.T + b)
// Block: 256 threads (8 waves). Block tile 128x64; each wave computes a
// 32x32 sub-tile as 2x2 accumulators -> 4 WMMAs per K-step per wave.
// Software-pipelined: double-buffered LDS K-tiles, next tile's global loads
// issued before the WMMAs so their latency overlaps matrix work; a single
// barrier per K-step.
// ---------------------------------------------------------------------------
__global__ __launch_bounds__(256) void gemm_bias_kernel(
    const float* __restrict__ X, const float* __restrict__ W,
    const float* __restrict__ bias, float* __restrict__ C,
    int M, int N, int K) {
  __shared__ __bf16 As[2][128][32];   // 16 KB
  __shared__ __bf16 Bs[2][64][32];    //  8 KB

  const int tid  = threadIdx.x;
  const int lane = tid & 31;
  const int wave = tid >> 5;
  const int lrow = lane & 15;
  const int lhi  = lane >> 4;          // 0 or 1 (wave32 half)

  const int mBase = blockIdx.y * 128;
  const int nBase = blockIdx.x * 64;
  const int mw = (wave >> 1) * 32;     // wave M offset in block tile
  const int nw = (wave & 1) * 32;      // wave N offset in block tile

  // cooperative loaders
  const int aRow = tid >> 1;           // 0..127
  const int aCol = (tid & 1) * 16;     // 0 / 16
  const int bRow = tid >> 2;           // 0..63
  const int bCol = (tid & 3) * 8;      // 0,8,16,24

  const float* xp = X + (long)(mBase + aRow) * K + aCol;
  const float* wp = W + (long)(nBase + bRow) * K + bCol;

  v8f acc00 = {}, acc01 = {}, acc10 = {}, acc11 = {};

  float aReg[16], bReg[8];

  // prologue: stage tile 0 into LDS[0]
#pragma unroll
  for (int j = 0; j < 16; ++j) aReg[j] = xp[j];
#pragma unroll
  for (int j = 0; j < 8; ++j)  bReg[j] = wp[j];
#pragma unroll
  for (int j = 0; j < 16; ++j) As[0][aRow][aCol + j] = (__bf16)aReg[j];
#pragma unroll
  for (int j = 0; j < 8; ++j)  Bs[0][bRow][bCol + j] = (__bf16)bReg[j];

  const int nTiles = K >> 5;
  int buf = 0;
  for (int t = 0; t < nTiles; ++t) {
    __syncthreads();   // LDS[buf] visible to all waves

    const bool more = (t + 1) < nTiles;
    const int  k1   = (t + 1) * 32;
    if (more) {
      __builtin_prefetch(xp + k1 + 32, 0, 1);   // L2 prefetch of tile t+2
      __builtin_prefetch(wp + k1 + 32, 0, 1);
      // issue next tile's global loads; no wait until after the WMMAs
#pragma unroll
      for (int j = 0; j < 16; ++j) aReg[j] = xp[k1 + j];
#pragma unroll
      for (int j = 0; j < 8; ++j)  bReg[j] = wp[k1 + j];
    }

    // fragments from LDS[buf]
    v16bf a0, a1, b0, b1;
    const int kb  = lhi * 8;
    const int kbb = lhi * 16;
#pragma unroll
    for (int e = 0; e < 8; ++e) {
      a0[e]     = As[buf][mw + lrow][kb + e];
      a0[e + 8] = As[buf][mw + lrow][kb + 16 + e];
      a1[e]     = As[buf][mw + 16 + lrow][kb + e];
      a1[e + 8] = As[buf][mw + 16 + lrow][kb + 16 + e];
    }
#pragma unroll
    for (int e = 0; e < 8; ++e) {
      b0[e]     = Bs[buf][nw + lrow][kb + e];
      b0[e + 8] = Bs[buf][nw + lrow][kb + 16 + e];
      b1[e]     = Bs[buf][nw + 16 + lrow][kb + e];
      b1[e + 8] = Bs[buf][nw + 16 + lrow][kb + 16 + e];
    }

    acc00 = wmma_bf16(a0, b0, acc00);
    acc01 = wmma_bf16(a0, b1, acc01);
    acc10 = wmma_bf16(a1, b0, acc10);
    acc11 = wmma_bf16(a1, b1, acc11);

    if (more) {          // store next tile into the other buffer
#pragma unroll
      for (int j = 0; j < 16; ++j) As[buf ^ 1][aRow][aCol + j] = (__bf16)aReg[j];
#pragma unroll
      for (int j = 0; j < 8; ++j)  Bs[buf ^ 1][bRow][bCol + j] = (__bf16)bReg[j];
    }
    buf ^= 1;
  }

  const int col0 = nBase + nw + lrow;
  const int col1 = col0 + 16;
  const float bv0 = bias ? bias[col0] : 0.0f;
  const float bv1 = bias ? bias[col1] : 0.0f;
#pragma unroll
  for (int v = 0; v < 8; ++v) {
    const int row0 = mBase + mw + v + lhi * 8;
    const int row1 = row0 + 16;
    C[(long)row0 * N + col0] = acc00[v] + bv0;
    C[(long)row0 * N + col1] = acc01[v] + bv1;
    C[(long)row1 * N + col0] = acc10[v] + bv0;
    C[(long)row1 * N + col1] = acc11[v] + bv1;
  }
}

// ---------------------------------------------------------------------------
// Flash attention: one wave (32 threads) per (b, h, 16-query tile).
// V tiles are double-buffered in LDS via async global->LDS copies so the
// next chunk's copy overlaps the current chunk's WMMAs + softmax.
// qkv layout: [4096][3072] f32  (q | k | v, head-major inside each third)
// out layout: [4096][1024] f32  (merged heads)
// ---------------------------------------------------------------------------
__global__ __launch_bounds__(32) void attn_kernel(
    const float* __restrict__ qkv, float* __restrict__ out) {
  __shared__ float  Vsf[2][32][HDIM];   // double-buffered V tile, f32, 16 KB
  __shared__ __bf16 Ps[16][32];         // P tile for layout transpose

  const int lane = threadIdx.x & 31;
  const int lrow = lane & 15;
  const int lhi  = lane >> 4;

  const int qt = blockIdx.x & 127;            // SEQ/16 = 128 query tiles
  const int bh = blockIdx.x >> 7;
  const int h  = bh & (NHEAD - 1);
  const int b  = bh >> 4;

  const int  qBase  = qt * 16;
  const long rowOff = (long)b * SEQ;
  const int  qOff   = h * HDIM;
  const int  kOff   = NEMBD + h * HDIM;
  const int  vOff   = 2 * NEMBD + h * HDIM;

  const unsigned vsBase = (unsigned)(uintptr_t)&Vsf[0][0][0]; // LDS byte base

  // Q fragments (16x64 -> two 16x32 A fragments), scale 1/sqrt(64) folded in.
  v16bf aq[2];
  {
    const float* qp = qkv + (rowOff + qBase + lrow) * LDQKV + qOff;
    const int kb = lhi * 8;
#pragma unroll
    for (int s = 0; s < 2; ++s) {
#pragma unroll
      for (int e = 0; e < 8; ++e) {
        aq[s][e]     = (__bf16)(qp[s * 32 + kb + e] * 0.125f);
        aq[s][e + 8] = (__bf16)(qp[s * 32 + 16 + kb + e] * 0.125f);
      }
    }
  }

  // issue async copy of a 32x64 f32 V tile into buffer `buf`
  auto issueV = [&](int kc, int buf) {
    const float* basep = qkv + (rowOff + kc) * LDQKV + vOff;
#pragma unroll
    for (int i = 0; i < 16; ++i) {
      const int lin = lane + i * 32;    // 0..511 b128 units
      const int r   = lin >> 4;         // key row 0..31
      const int c4  = lin & 15;         // float4 column
      const unsigned off = vsBase + ((unsigned)buf * 32 * HDIM +
                                     (unsigned)r * HDIM + c4 * 4) * 4u;
      async_ld128(off, basep + (long)r * LDQKV + c4 * 4);
    }
  };

  v8f o0 = {}, o1 = {}, o2 = {}, o3 = {};
  float mrow[8], lsum[8];
#pragma unroll
  for (int v = 0; v < 8; ++v) { mrow[v] = -3.0e38f; lsum[v] = 0.0f; }

  const int kcEnd = qBase + 16;               // causal: only keys <= last query
  issueV(0, 0);
  int buf = 0;

  for (int kc = 0; kc < kcEnd; kc += 32) {
    const bool haveNext = (kc + 32) < kcEnd;
    if (haveNext) issueV(kc + 32, buf ^ 1);   // overlap next copy with compute
    if (haveNext) wait_async_16();            // current tile (16 ops) done
    else          wait_async_0();
    __syncthreads();

    // scores: two 16x16 tiles, each accumulating over D=64 in two K steps
    v8f sc[2];
#pragma unroll
    for (int t = 0; t < 2; ++t) {
      v8f c = {};
#pragma unroll
      for (int s = 0; s < 2; ++s) {
        v16bf bk;   // B fragment of K^T: lane = key column, d in two runs of 8 (table order)
        const float* kp = qkv + (rowOff + kc + t * 16 + lrow) * LDQKV + kOff +
                          s * 32 + lhi * 8;
#pragma unroll
        for (int e = 0; e < 8; ++e) { bk[e] = (__bf16)kp[e]; bk[e + 8] = (__bf16)kp[16 + e]; }
        c = wmma_bf16(aq[s], bk, c);
      }
      sc[t] = c;
    }

    // causal mask (C layout: row = qBase + v + 8*lhi, key = kc + t*16 + lrow)
#pragma unroll
    for (int v = 0; v < 8; ++v) {
      const int qrow = qBase + v + lhi * 8;
      if (kc + lrow      > qrow) sc[0][v] = NEGV;
      if (kc + 16 + lrow > qrow) sc[1][v] = NEGV;
    }

    // online softmax (row reductions across the 16-lane half via shfl_xor)
    float alpha[8];
#pragma unroll
    for (int v = 0; v < 8; ++v) {
      float x = fmaxf(sc[0][v], sc[1][v]);
#pragma unroll
      for (int off = 8; off >= 1; off >>= 1)
        x = fmaxf(x, __shfl_xor(x, off, 16));
      const float mnew = fmaxf(mrow[v], x);
      alpha[v] = __expf(mrow[v] - mnew);
      mrow[v]  = mnew;
      const float p0 = __expf(sc[0][v] - mnew);
      const float p1 = __expf(sc[1][v] - mnew);
      sc[0][v] = p0; sc[1][v] = p1;
      float s2 = p0 + p1;
#pragma unroll
      for (int off = 8; off >= 1; off >>= 1)
        s2 += __shfl_xor(s2, off, 16);
      lsum[v] = lsum[v] * alpha[v] + s2;
      o0[v] *= alpha[v]; o1[v] *= alpha[v];
      o2[v] *= alpha[v]; o3[v] *= alpha[v];
    }

    // transpose P (C layout -> A layout) through LDS
#pragma unroll
    for (int v = 0; v < 8; ++v) {
      Ps[v + lhi * 8][lrow]      = (__bf16)sc[0][v];
      Ps[v + lhi * 8][16 + lrow] = (__bf16)sc[1][v];
    }
    __syncthreads();

    v16bf ap;
    {
      const int kb = lhi * 8;
#pragma unroll
      for (int e = 0; e < 8; ++e) {
        ap[e]     = Ps[lrow][kb + e];
        ap[e + 8] = Ps[lrow][kb + 16 + e];
      }
    }

    // O += P (16x32) x V (32x64): four 16x16 output columns
#pragma unroll
    for (int j = 0; j < 4; ++j) {
      v16bf bv;
#pragma unroll
      for (int e = 0; e < 8; ++e) {
        bv[e]     = (__bf16)Vsf[buf][lhi * 8 + e][j * 16 + lrow];
        bv[e + 8] = (__bf16)Vsf[buf][lhi * 8 + 16 + e][j * 16 + lrow];
      }
      if      (j == 0) o0 = wmma_bf16(ap, bv, o0);
      else if (j == 1) o1 = wmma_bf16(ap, bv, o1);
      else if (j == 2) o2 = wmma_bf16(ap, bv, o2);
      else             o3 = wmma_bf16(ap, bv, o3);
    }
    __syncthreads();
    buf ^= 1;
  }

  // normalize and store merged-head output
#pragma unroll
  for (int v = 0; v < 8; ++v) {
    const int qrow = qBase + v + lhi * 8;
    const float inv = 1.0f / lsum[v];
    const long base = (rowOff + qrow) * (long)NEMBD + h * HDIM;
    out[base + 0  + lrow] = o0[v] * inv;
    out[base + 16 + lrow] = o1[v] * inv;
    out[base + 32 + lrow] = o2[v] * inv;
    out[base + 48 + lrow] = o3[v] * inv;
  }
}

// ---------------------------------------------------------------------------
extern "C" void kernel_launch(void* const* d_in, const int* in_sizes, int n_in,
                              void* d_out, int out_size, void* d_ws, size_t ws_size,
                              hipStream_t stream) {
  const float* hs     = (const float*)d_in[0];   // [2,2048,1024]
  const float* w_attn = (const float*)d_in[1];   // [3072,1024]
  const float* b_attn = (const float*)d_in[2];   // [3072]
  const float* w_proj = (const float*)d_in[3];   // [1024,1024]
  const float* b_proj = (const float*)d_in[4];   // [1024]
  float* out = (float*)d_out;                    // [2,2048,1024]

  float* qkv  = (float*)d_ws;                          // 4096*3072 f32
  float* attn = qkv + (size_t)MROWS * LDQKV;           // 4096*1024 f32

  // 1) fused QKV projection
  dim3 g1(LDQKV / 64, MROWS / 128);
  gemm_bias_kernel<<<g1, 256, 0, stream>>>(hs, w_attn, b_attn, qkv,
                                           MROWS, LDQKV, NEMBD);
  // 2) causal flash attention (one wave per (b,h,q-tile))
  attn_kernel<<<dim3(2 * NHEAD * (SEQ / 16)), 32, 0, stream>>>(qkv, attn);
  // 3) output projection
  dim3 g3(NEMBD / 64, MROWS / 128);
  gemm_bias_kernel<<<g3, 256, 0, stream>>>(attn, w_proj, b_proj, out,
                                           MROWS, NEMBD, NEMBD);
}
